// HybridGroupedSpatialMambaModel2_51067161150076
// MI455X (gfx1250) — matrix operand order repaired and checked by reading, not compiled, not run on hardware
//
#include <hip/hip_runtime.h>
#include <hip/hip_bf16.h>

// ---------------------------------------------------------------------------
// Model constants (from the reference)
// ---------------------------------------------------------------------------
constexpr int BATCH  = 1024;
constexpr int SEQ    = 256;
constexpr int FSP    = 32;
constexpr int DSP    = 64;
constexpr int DIN    = 128;
constexpr int DSTATE = 16;
constexpr int DCONV  = 4;
constexpr int NLAY   = 2;
constexpr size_t MROWS = (size_t)BATCH * SEQ;   // 262144 GEMM rows

typedef __attribute__((ext_vector_type(16))) _Float16 v16h;
typedef __attribute__((ext_vector_type(8)))  _Float16 v8h;
typedef __attribute__((ext_vector_type(8)))  float    v8f;

// ---------------------------------------------------------------------------
// WMMA fragment helpers (layouts per CDNA5 ISA 7.12.2, wave32)
// A 16x32 f16: lane(0-15)=row M; halves 0..7 -> K +0..7 (+8 for lanes>=16),
//              halves 8..15 -> K +16..23 (+8 for lanes>=16).
// ---------------------------------------------------------------------------
__device__ inline v16h load_a_frag(const _Float16* __restrict__ A, int lda,
                                   int mbase, int kbase, int lane) {
  const _Float16* row = A + (size_t)(mbase + (lane & 15)) * lda
                          + kbase + ((lane >> 4) << 3);
  v8h lo = *(const v8h*)(row);
  v8h hi = *(const v8h*)(row + 16);
  return __builtin_shufflevector(lo, hi, 0,1,2,3,4,5,6,7,8,9,10,11,12,13,14,15);
}

// Pre-packed B fragment: one contiguous v16h per (tile, lane).
__device__ inline v16h load_b_packed(const _Float16* __restrict__ Wp,
                                     int tile, int lane) {
  return *(const v16h*)(Wp + ((size_t)tile * 32 + lane) * 16);
}

// ---------------------------------------------------------------------------
// Convert + pack weights (K x N, f32 row-major) into WMMA B-fragment order:
// dst[((kt*ntiles + nt)*32 + lane)*16 + h] = W[k, n],
//   n = nt*16 + (lane&15),  k = kt*32 + ((lane>>4)<<3) + (h&7) + ((h>>3)<<4)
// ---------------------------------------------------------------------------
__global__ void pack_b_kernel(const float* __restrict__ src,
                              _Float16* __restrict__ dst,
                              int Kd, int N) {
  int o = blockIdx.x * 256 + threadIdx.x;
  if (o >= Kd * N) return;
  int h    = o & 15;
  int lane = (o >> 4) & 31;
  int tile = o >> 9;
  int ntiles = N >> 4;
  int nt = tile % ntiles;
  int kt = tile / ntiles;
  int n = nt * 16 + (lane & 15);
  int k = kt * 32 + ((lane >> 4) << 3) + (h & 7) + ((h >> 3) << 4);
  dst[o] = (_Float16)src[(size_t)k * N + n];
}

__global__ void cvt_f32_f16_kernel(const float* __restrict__ src,
                                   _Float16* __restrict__ dst, int n) {
  int i = blockIdx.x * 256 + threadIdx.x;
  if (i < n) dst[i] = (_Float16)src[i];
}

// ---------------------------------------------------------------------------
// WMMA GEMM: C(M x NTILES*16) = A_f16(M x KD) * Wpacked (+bias).
// One wave owns 16 rows x (NGROUP*16) cols: A fragment reused NGROUP times.
// All shape/output options are compile-time -> branch-free epilogue.
// ---------------------------------------------------------------------------
template <int KD, int NTILES, int NGROUP, bool BIAS, bool OUTF32, bool OUTF16>
__global__ void wmma_gemm_kernel(const _Float16* __restrict__ A,
                                 const _Float16* __restrict__ Wp,
                                 const float* __restrict__ bias,
                                 float* __restrict__ Cf,
                                 _Float16* __restrict__ Ch) {
  constexpr int N = NTILES * 16;
  constexpr int NGRP = NTILES / NGROUP;
  int lane = threadIdx.x & 31;
  int wave = threadIdx.x >> 5;
  int id   = blockIdx.x * (blockDim.x >> 5) + wave;
  int mt   = id / NGRP;
  int ng   = id - mt * NGRP;
  int nt0  = ng * NGROUP;

  v8f acc[NGROUP];
#pragma unroll
  for (int g = 0; g < NGROUP; ++g) acc[g] = (v8f){};

#pragma unroll
  for (int kt = 0; kt < KD / 32; ++kt) {
    v16h a = load_a_frag(A, KD, mt * 16, kt * 32, lane);
#pragma unroll
    for (int g = 0; g < NGROUP; ++g) {
      v16h b = load_b_packed(Wp, kt * NTILES + nt0 + g, lane);
      acc[g] = __builtin_amdgcn_wmma_f32_16x16x32_f16(
          false, a, false, b, (short)0, acc[g], false, false);
    }
  }

  int m0 = mt * 16 + ((lane >> 4) << 3);
#pragma unroll
  for (int g = 0; g < NGROUP; ++g) {
    int n = (nt0 + g) * 16 + (lane & 15);
    float bn = BIAS ? bias[n] : 0.f;
#pragma unroll
    for (int r = 0; r < 8; ++r) {
      float v = acc[g][r] + bn;
      size_t idx = (size_t)(m0 + r) * N + n;
      if constexpr (OUTF32) Cf[idx] = v;
      if constexpr (OUTF16) Ch[idx] = (_Float16)v;
    }
  }
}

// ---------------------------------------------------------------------------
// Grouped tiny MLPs + LayerNorm(32) — one wave32 per row, shfl reductions.
// ---------------------------------------------------------------------------
struct GroupParams {
  const float* w[4]; const float* b[4]; const float* g[4]; const float* be[4];
};

__global__ void group_kernel(const float* __restrict__ xf, GroupParams P,
                             float* __restrict__ out) {
  int lane = threadIdx.x & 31;
  int wave = threadIdx.x >> 5;
  int row  = blockIdx.x * 8 + wave;
  float xr[14];
#pragma unroll
  for (int i = 0; i < 14; ++i) xr[i] = xf[(size_t)row * 14 + i];
  const int off[4] = {0, 4, 7, 11};
  const int dim[4] = {4, 3, 4, 3};
#pragma unroll
  for (int gi = 0; gi < 4; ++gi) {
    float h = P.b[gi][lane];
    for (int d = 0; d < dim[gi]; ++d)
      h = fmaf(xr[off[gi] + d], P.w[gi][d * 32 + lane], h);
    h = fmaxf(h, 0.f);
    float s = h;
    for (int m = 16; m; m >>= 1) s += __shfl_xor(s, m, 32);
    float mean = s * (1.f / 32.f);
    float dv = h - mean, v = dv * dv;
    for (int m = 16; m; m >>= 1) v += __shfl_xor(v, m, 32);
    float rs = rsqrtf(v * (1.f / 32.f) + 1e-5f);
    out[(size_t)row * 128 + gi * 32 + lane] = dv * rs * P.g[gi][lane] + P.be[gi][lane];
  }
}

// ---------------------------------------------------------------------------
// Depthwise causal conv(4) + bias + SiLU. One thread per (b,t,d).
// ---------------------------------------------------------------------------
__global__ void conv_silu_kernel(const _Float16* __restrict__ XZ,
                                 const float* __restrict__ cw,
                                 const float* __restrict__ cb,
                                 float* __restrict__ XC) {
  size_t idx = (size_t)blockIdx.x * 256 + threadIdx.x;  // over MROWS*DIN
  int d = (int)(idx & (DIN - 1));
  size_t bt = idx >> 7;
  int t = (int)(bt & (SEQ - 1));
  size_t brow = bt - t;
  float acc = cb[d];
#pragma unroll
  for (int j = 0; j < DCONV; ++j) {
    int tt = t - (DCONV - 1) + j;
    float xv = (tt >= 0) ? (float)XZ[(brow + tt) * (2 * DIN) + d] : 0.f;
    acc = fmaf(cw[d * DCONV + j], xv, acc);
  }
  XC[idx] = acc / (1.f + __expf(-acc));
}

// ---------------------------------------------------------------------------
// x-proj (128->36), split into dt_raw/B/C, dt = softplus(dt_raw@dt_w + dt_b).
// 8 rows per block, LDS staging.
// ---------------------------------------------------------------------------
__global__ void xproj_kernel(const float* __restrict__ XC,
                             const float* __restrict__ xpw,   // 128x36
                             const float* __restrict__ dtw,   // 4x128
                             const float* __restrict__ dtb,   // 128
                             float* __restrict__ DT,
                             float* __restrict__ BM,
                             float* __restrict__ CM) {
  __shared__ float xcl[8][DIN];
  __shared__ float xdl[8][36];
  size_t row0 = (size_t)blockIdx.x * 8;
  for (int i = threadIdx.x; i < 8 * DIN; i += 256)
    xcl[i >> 7][i & 127] = XC[row0 * DIN + i];
  __syncthreads();
  for (int o = threadIdx.x; o < 8 * 36; o += 256) {
    int r = o / 36, c = o - r * 36;
    float s = 0.f;
    for (int k = 0; k < DIN; ++k) s = fmaf(xcl[r][k], xpw[k * 36 + c], s);
    xdl[r][c] = s;
    if (c >= 4 && c < 20)      BM[(row0 + r) * DSTATE + (c - 4)]  = s;
    else if (c >= 20)          CM[(row0 + r) * DSTATE + (c - 20)] = s;
  }
  __syncthreads();
  for (int o = threadIdx.x; o < 8 * DIN; o += 256) {
    int r = o >> 7, d = o & 127;
    float s = dtb[d];
#pragma unroll
    for (int j = 0; j < 4; ++j) s = fmaf(xdl[r][j], dtw[j * DIN + d], s);
    DT[(row0 + r) * DIN + d] = (s > 20.f) ? s : log1pf(__expf(s));
  }
}

// ---------------------------------------------------------------------------
// Selective scan: one block per batch element, one lane per channel d.
// h[16] lives in registers; 256 sequential steps; gated output -> f16.
// ---------------------------------------------------------------------------
__global__ void scan_kernel(const float* __restrict__ XC,
                            const float* __restrict__ DT,
                            const float* __restrict__ BM,
                            const float* __restrict__ CM,
                            const _Float16* __restrict__ XZ,  // z at cols 128..255
                            const float* __restrict__ Alog,
                            const float* __restrict__ Dp,
                            _Float16* __restrict__ YH) {
  int d = threadIdx.x;
  int b = blockIdx.x;
  float Ar[DSTATE];
#pragma unroll
  for (int s = 0; s < DSTATE; ++s) Ar[s] = -__expf(Alog[d * DSTATE + s]);
  float Dd = Dp[d];
  float h[DSTATE];
#pragma unroll
  for (int s = 0; s < DSTATE; ++s) h[s] = 0.f;

  size_t base = (size_t)b * SEQ;
  for (int t = 0; t < SEQ; ++t) {
    size_t row = base + t;
    if (t + 4 < SEQ) {
      __builtin_prefetch(DT + (row + 4) * DIN + d, 0, 1);   // global_prefetch_b8
      __builtin_prefetch(XC + (row + 4) * DIN + d, 0, 1);
    }
    float dt = DT[row * DIN + d];
    float xt = XC[row * DIN + d];
    float zt = (float)XZ[row * (2 * DIN) + DIN + d];
    const float4* Bp = (const float4*)(BM + row * DSTATE);
    const float4* Cp = (const float4*)(CM + row * DSTATE);
    float dtx = dt * xt;
    float y = 0.f;
#pragma unroll
    for (int q = 0; q < 4; ++q) {
      float4 Bv = Bp[q], Cv = Cp[q];
      float bB[4] = {Bv.x, Bv.y, Bv.z, Bv.w};
      float cC[4] = {Cv.x, Cv.y, Cv.z, Cv.w};
#pragma unroll
      for (int e = 0; e < 4; ++e) {
        int s = q * 4 + e;
        float dA = __expf(dt * Ar[s]);
        h[s] = fmaf(dA, h[s], dtx * bB[e]);
        y = fmaf(h[s], cC[e], y);
      }
    }
    float yv = (y + xt * Dd) * (zt / (1.f + __expf(-zt)));
    YH[row * DIN + d] = (_Float16)yv;
  }
}

// ---------------------------------------------------------------------------
// Out-proj WMMA (K=128, N=64) fused with LayerNorm(64) + residual.
// Block = 128 threads (4 waves) -> one 16-row strip, full 64 columns.
// ---------------------------------------------------------------------------
__global__ void outproj_ln_kernel(const _Float16* __restrict__ Yh,
                                  const _Float16* __restrict__ Wp,  // packed frags
                                  const float* __restrict__ res,
                                  const float* __restrict__ lng,
                                  const float* __restrict__ lnb,
                                  float* __restrict__ Xo,
                                  _Float16* __restrict__ Xoh) {
  __shared__ float tile[16 * 65];
  __shared__ float mu[16], rs[16];
  int lane = threadIdx.x & 31;
  int wave = threadIdx.x >> 5;       // 4 waves -> N tiles 0..3
  int mbase = blockIdx.x * 16;

  v8f acc = {};
#pragma unroll
  for (int kt = 0; kt < DIN / 32; ++kt) {
    v16h a = load_a_frag(Yh, DIN, mbase, kt * 32, lane);
    v16h b = load_b_packed(Wp, kt * (DSP / 16) + wave, lane);
    acc = __builtin_amdgcn_wmma_f32_16x16x32_f16(
        false, a, false, b, (short)0, acc, false, false);
  }
  int n  = wave * 16 + (lane & 15);
  int m0 = (lane >> 4) << 3;
#pragma unroll
  for (int r = 0; r < 8; ++r) tile[(m0 + r) * 65 + n] = acc[r];
  __syncthreads();

  if (threadIdx.x < 16) {
    int m = threadIdx.x;
    float s = 0.f;
    for (int j = 0; j < DSP; ++j) s += tile[m * 65 + j];
    float mean = s * (1.f / DSP);
    float v = 0.f;
    for (int j = 0; j < DSP; ++j) {
      float dv = tile[m * 65 + j] - mean;
      v = fmaf(dv, dv, v);
    }
    mu[m] = mean;
    rs[m] = rsqrtf(v * (1.f / DSP) + 1e-5f);
  }
  __syncthreads();

  for (int idx = threadIdx.x; idx < 16 * DSP; idx += 128) {
    int m = idx >> 6, c = idx & 63;
    size_t gi = (size_t)(mbase + m) * DSP + c;
    float v = (tile[m * 65 + c] - mu[m]) * rs[m] * lng[c] + lnb[c] + res[gi];
    Xo[gi]  = v;
    Xoh[gi] = (_Float16)v;
  }
}

// ---------------------------------------------------------------------------
// Mean over SEQ + head MLP (192->64 relu ->1) + sigmoid. One block per batch.
// ---------------------------------------------------------------------------
__global__ void head_kernel(const float* __restrict__ X,
                            const float* __restrict__ G,
                            const float* __restrict__ h1w,  // 192x64
                            const float* __restrict__ h1b,
                            const float* __restrict__ h2w,  // 64
                            const float* __restrict__ h2b,
                            float* __restrict__ out) {
  __shared__ float part[4][64];
  __shared__ float sps[64];
  __shared__ float red[64];
  int b = blockIdx.x, t = threadIdx.x;
  int n = t & 63, ch = t >> 6;

  size_t base = (size_t)b * SEQ * DSP;
  float s = 0.f;
  for (int k = ch * 64; k < ch * 64 + 64; ++k) s += X[base + (size_t)k * DSP + n];
  part[ch][n] = s;
  __syncthreads();
  if (t < 64) sps[t] = (part[0][t] + part[1][t] + part[2][t] + part[3][t]) * (1.f / SEQ);
  __syncthreads();
  if (t < 64) {
    float a = h1b[t];
    const float* gr = G + (size_t)b * 128;
    for (int j = 0; j < 128; ++j) a = fmaf(gr[j], h1w[j * 64 + t], a);
    for (int j = 0; j < 64; ++j)  a = fmaf(sps[j], h1w[(128 + j) * 64 + t], a);
    red[t] = fmaxf(a, 0.f) * h2w[t];
  }
  __syncthreads();
  for (int st = 32; st > 0; st >>= 1) {
    if (t < st) red[t] += red[t + st];
    __syncthreads();
  }
  if (t == 0) out[b] = 1.f / (1.f + __expf(-(red[0] + h2b[0])));
}

// ---------------------------------------------------------------------------
// Launch
// ---------------------------------------------------------------------------
extern "C" void kernel_launch(void* const* d_in, const int* in_sizes, int n_in,
                              void* d_out, int out_size, void* d_ws, size_t ws_size,
                              hipStream_t stream) {
  const float* x_flat    = (const float*)d_in[0];
  const float* x_spatial = (const float*)d_in[1];
  GroupParams gp;
  for (int i = 0; i < 4; ++i) {
    gp.w[i]  = (const float*)d_in[2 + i * 4 + 0];
    gp.b[i]  = (const float*)d_in[2 + i * 4 + 1];
    gp.g[i]  = (const float*)d_in[2 + i * 4 + 2];
    gp.be[i] = (const float*)d_in[2 + i * 4 + 3];
  }
  const float* sp_w  = (const float*)d_in[18];
  const float* sp_b  = (const float*)d_in[19];
  const float* in_w  = (const float*)d_in[20];
  const float* convw = (const float*)d_in[21];
  const float* convb = (const float*)d_in[22];
  const float* xp_w  = (const float*)d_in[23];
  const float* dt_w  = (const float*)d_in[24];
  const float* dt_b  = (const float*)d_in[25];
  const float* A_log = (const float*)d_in[26];
  const float* D_    = (const float*)d_in[27];
  const float* out_w = (const float*)d_in[28];
  const float* ln_g  = (const float*)d_in[29];
  const float* ln_b  = (const float*)d_in[30];
  const float* h1_w  = (const float*)d_in[31];
  const float* h1_b  = (const float*)d_in[32];
  const float* h2_w  = (const float*)d_in[33];
  const float* h2_b  = (const float*)d_in[34];
  float* out = (float*)d_out;

  // --- workspace layout (256B aligned) ---
  char* ws = (char*)d_ws;
  size_t off = 0;
  auto take = [&](size_t bytes) {
    size_t r = off;
    off += (bytes + 255) & ~(size_t)255;
    return r;
  };
  size_t oXSH = take(MROWS * FSP * 2);                 // x_spatial f16
  size_t oSPW = take((size_t)FSP * DSP * 2);           // sp_w packed frags
  size_t oINW = take((size_t)NLAY * DSP * 2 * DIN * 2);// in_w packed frags
  size_t oOUW = take((size_t)NLAY * DIN * DSP * 2);    // out_w packed frags
  size_t oX0  = take(MROWS * DSP * 4);
  size_t oX1  = take(MROWS * DSP * 4);
  size_t oX0H = take(MROWS * DSP * 2);
  size_t oX1H = take(MROWS * DSP * 2);
  size_t oXZ  = take(MROWS * 2 * DIN * 2);             // xz f16 (xc_raw | z)
  size_t oXC  = take(MROWS * DIN * 4);
  size_t oDT  = take(MROWS * DIN * 4);
  size_t oBM  = take(MROWS * DSTATE * 4);
  size_t oCM  = take(MROWS * DSTATE * 4);
  size_t oYH  = take(MROWS * DIN * 2);
  size_t oGRP = take((size_t)BATCH * 128 * 4);
  (void)ws_size; (void)n_in; (void)in_sizes; (void)out_size;

  _Float16* XSH = (_Float16*)(ws + oXSH);
  _Float16* SPW = (_Float16*)(ws + oSPW);
  _Float16* INW = (_Float16*)(ws + oINW);
  _Float16* OUW = (_Float16*)(ws + oOUW);
  float*    X0  = (float*)(ws + oX0);
  float*    X1  = (float*)(ws + oX1);
  _Float16* X0H = (_Float16*)(ws + oX0H);
  _Float16* X1H = (_Float16*)(ws + oX1H);
  _Float16* XZ  = (_Float16*)(ws + oXZ);
  float*    XC  = (float*)(ws + oXC);
  float*    DT  = (float*)(ws + oDT);
  float*    BM  = (float*)(ws + oBM);
  float*    CM  = (float*)(ws + oCM);
  _Float16* YH  = (_Float16*)(ws + oYH);
  float*    GRP = (float*)(ws + oGRP);

  // --- convert A-side activations, pack all weight matrices as B fragments ---
  {
    int n = (int)(MROWS * FSP);
    cvt_f32_f16_kernel<<<(n + 255) / 256, 256, 0, stream>>>(x_spatial, XSH, n);
    pack_b_kernel<<<(FSP * DSP + 255) / 256, 256, 0, stream>>>(sp_w, SPW, FSP, DSP);
    for (int l = 0; l < NLAY; ++l) {
      pack_b_kernel<<<(DSP * 2 * DIN + 255) / 256, 256, 0, stream>>>(
          in_w + (size_t)l * DSP * 2 * DIN, INW + (size_t)l * DSP * 2 * DIN,
          DSP, 2 * DIN);
      pack_b_kernel<<<(DIN * DSP + 255) / 256, 256, 0, stream>>>(
          out_w + (size_t)l * DIN * DSP, OUW + (size_t)l * DIN * DSP,
          DIN, DSP);
    }
  }

  // --- grouped MLPs ---
  group_kernel<<<BATCH / 8, 256, 0, stream>>>(x_flat, gp, GRP);

  // --- spatial projection: X0 = x_spatial @ sp_w + sp_b  (WMMA) ---
  {
    // KD=32, N=64 (4 tiles, all 4 per wave), bias, f32+f16 out
    int waves = (int)(MROWS / 16);            // 16384 waves
    wmma_gemm_kernel<32, 4, 4, true, true, true>
        <<<waves / 8, 256, 0, stream>>>(XSH, SPW, sp_b, X0, X0H);
  }

  // --- Mamba layers ---
  for (int l = 0; l < NLAY; ++l) {
    const _Float16* XinH = (l == 0) ? X0H : X1H;
    const float*    Xres = (l == 0) ? X0 : X1;
    float*          Xout = (l == 0) ? X1 : X0;
    _Float16*       XoutH= (l == 0) ? X1H : X0H;

    // in-proj: xz = x @ in_w[l]   (M x 64) @ (64 x 256) -> f16
    {
      // KD=64, N=256 (16 tiles, 4 per wave), no bias, f16 out
      int waves = (int)(MROWS / 16) * 4;      // 65536 waves
      wmma_gemm_kernel<64, 16, 4, false, false, true>
          <<<waves / 8, 256, 0, stream>>>(
              XinH, INW + (size_t)l * DSP * 2 * DIN, nullptr, nullptr, XZ);
    }
    // causal depthwise conv + SiLU
    conv_silu_kernel<<<(int)(MROWS * DIN / 256), 256, 0, stream>>>(
        XZ, convw + (size_t)l * DIN * DCONV, convb + (size_t)l * DIN, XC);
    // x-proj + dt/B/C
    xproj_kernel<<<(int)(MROWS / 8), 256, 0, stream>>>(
        XC, xp_w + (size_t)l * DIN * 36, dt_w + (size_t)l * 4 * DIN,
        dt_b + (size_t)l * DIN, DT, BM, CM);
    // selective scan (+ gate by silu(z)) -> y f16
    scan_kernel<<<BATCH, DIN, 0, stream>>>(
        XC, DT, BM, CM, XZ, A_log + (size_t)l * DIN * DSTATE,
        D_ + (size_t)l * DIN, YH);
    // out-proj + LN + residual (WMMA fused)
    outproj_ln_kernel<<<(int)(MROWS / 16), 128, 0, stream>>>(
        YH, OUW + (size_t)l * DIN * DSP, Xres,
        ln_g + (size_t)l * DSP, ln_b + (size_t)l * DSP, Xout, XoutH);
  }

  // --- mean pool + head (final x lives in X0 after 2 layers) ---
  head_kernel<<<BATCH, 256, 0, stream>>>(X0, GRP, h1_w, h1_b, h2_w, h2_b, out);
}
